// GQAAttention_21947282882825
// MI455X (gfx1250) — compile-verified, matrix operand order checked by reading, not run on hardware
//
#include <hip/hip_runtime.h>
#include <math.h>

// ---------------------------------------------------------------------------
// GQA attention forward for MI455X (gfx1250, wave32, WMMA).
// B=2, T=2048, D=4096, H=32, G=8, d=128. All matmuls on
// v_wmma_f32_16x16x32_bf16 (bf16 in, fp32 accumulate).
// ---------------------------------------------------------------------------

typedef unsigned short u16;
typedef __attribute__((ext_vector_type(16))) unsigned short v16u;
typedef __attribute__((ext_vector_type(8)))  unsigned short v8u;
typedef __attribute__((ext_vector_type(16))) __bf16        v16bf;
typedef __attribute__((ext_vector_type(8)))  float         v8f;

#define B_ 2
#define T_ 2048
#define D_ 4096
#define H_ 32
#define G_ 8
#define d_ 128
#define KV_ 1024  // G_*d_

union FragU { v16u u; v16bf b; };

__device__ inline v8f wmma_bf16(v16u a, v16u b, v8f c) {
  FragU fa; fa.u = a;
  FragU fb; fb.u = b;
  return __builtin_amdgcn_wmma_f32_16x16x32_bf16(
      /*neg_a=*/false, fa.b, /*neg_b=*/false, fb.b,
      /*c_mod=*/(short)0, c, /*reuse_a=*/false, /*reuse_b=*/false);
}

__device__ inline u16 f2bf(float f) {
  union { float f; unsigned u; } v; v.f = f;
  unsigned r = v.u + 0x7FFFu + ((v.u >> 16) & 1u);  // round-to-nearest-even
  return (u16)(r >> 16);
}

__device__ inline v8f vzero8() {
  v8f v;
#pragma unroll
  for (int i = 0; i < 8; ++i) v[i] = 0.0f;
  return v;
}

// Fragment loader for v_wmma_f32_16x16x32_bf16: 16 rows x 32 cols, row-major,
// K-contiguous source (works for A tiles and for B tiles stored as N rows x
// K cols). lane&15 -> row; per lane two contiguous 8xbf16 (16B) chunks.
__device__ inline v16u load_frag_16x32(const u16* p, int stride) {
  const int lane = threadIdx.x & 31;
  const int m = lane & 15;
  const int hh = lane >> 4;
  const u16* r = p + (size_t)m * stride;
  v8u lo = *(const v8u*)(r + hh * 8);
  v8u hi = *(const v8u*)(r + 16 + hh * 8);
  v16u out;
#pragma unroll
  for (int i = 0; i < 8; ++i) { out[i] = lo[i]; out[8 + i] = hi[i]; }
  return out;
}

__device__ inline float rowmax16(float v) {
#pragma unroll
  for (int m = 1; m <= 8; m <<= 1) v = fmaxf(v, __shfl_xor(v, m, 32));
  return v;
}
__device__ inline float rowsum16(float v) {
#pragma unroll
  for (int m = 1; m <= 8; m <<= 1) v += __shfl_xor(v, m, 32);
  return v;
}

// ---------------------------------------------------------------------------
// fp32 -> bf16 convert
// ---------------------------------------------------------------------------
__global__ __launch_bounds__(256) void k_f32_to_bf16(const float* __restrict__ in,
                                                     u16* __restrict__ out,
                                                     size_t n) {
  size_t i = (size_t)blockIdx.x * blockDim.x + threadIdx.x;
  size_t st = (size_t)gridDim.x * blockDim.x;
  for (; i < n; i += st) out[i] = f2bf(in[i]);
}

// ---------------------------------------------------------------------------
// Y[M,N] = alpha * (X[M,K] * W[N,K]^T + bias[N]); X,W bf16, acc fp32.
// Block tile 256x128, K sliced by 32 through LDS. 8 waves in a 4x2 grid,
// each wave owns a 64x64 tile (4x4 WMMA frags): 16 WMMA per K-slice with
// only 8 fragment loads -> wmma-bound inner loop.
// ---------------------------------------------------------------------------
#define GT_M 256
#define GT_N 128
#define GT_K 32
#define LDSTR 40  // 32 + 8 pad, keeps 16B alignment

__global__ __launch_bounds__(256) void k_gemm_xwt(const u16* __restrict__ X,
                                                  const u16* __restrict__ W,
                                                  const float* __restrict__ bias,
                                                  void* __restrict__ out,
                                                  int M, int N, int K,
                                                  float alpha, int out_bf16) {
  __shared__ u16 Xs[GT_M * LDSTR];
  __shared__ u16 Ws[GT_N * LDSTR];

  const int tid  = threadIdx.x;
  const int lane = tid & 31;
  const int w    = tid >> 5;          // 0..7
  const int wm   = (w >> 1) * 64;     // 4 wave-rows
  const int wn   = (w & 1) * 64;      // 2 wave-cols
  const int bm   = blockIdx.y * GT_M;
  const int bn   = blockIdx.x * GT_N;

  v8f acc[4][4];
#pragma unroll
  for (int i = 0; i < 4; ++i)
#pragma unroll
    for (int j = 0; j < 4; ++j) acc[i][j] = vzero8();

  for (int kk = 0; kk < K; kk += GT_K) {
    // Cooperative staging: X 256x32, W 128x32 (16B chunks).
#pragma unroll
    for (int j = 0; j < 4; ++j) {
      int idx = tid + j * 256;        // 0..1023
      int row = idx >> 2;
      int seg = (idx & 3) * 8;
      const v8u* sx = (const v8u*)(X + (size_t)(bm + row) * K + kk + seg);
      *(v8u*)(Xs + row * LDSTR + seg) = *sx;
      if (kk + GT_K < K)
        __builtin_prefetch((const char*)sx + GT_K * sizeof(u16), 0, 1);
    }
#pragma unroll
    for (int j = 0; j < 2; ++j) {
      int idx = tid + j * 256;        // 0..511
      int row = idx >> 2;
      int seg = (idx & 3) * 8;
      const v8u* sw = (const v8u*)(W + (size_t)(bn + row) * K + kk + seg);
      *(v8u*)(Ws + row * LDSTR + seg) = *sw;
      if (kk + GT_K < K)
        __builtin_prefetch((const char*)sw + GT_K * sizeof(u16), 0, 1);
    }
    __syncthreads();

    v16u a[4], b[4];
#pragma unroll
    for (int mt = 0; mt < 4; ++mt)
      a[mt] = load_frag_16x32(Xs + (wm + mt * 16) * LDSTR, LDSTR);
#pragma unroll
    for (int nt = 0; nt < 4; ++nt)
      b[nt] = load_frag_16x32(Ws + (wn + nt * 16) * LDSTR, LDSTR);

#pragma unroll
    for (int mt = 0; mt < 4; ++mt)
#pragma unroll
      for (int nt = 0; nt < 4; ++nt)
        acc[mt][nt] = wmma_bf16(a[mt], b[nt], acc[mt][nt]);

    __syncthreads();
  }

  // Epilogue: C layout = lane&15 -> col, vgpr i -> row i + 8*(lane>>4).
  const int hh = lane >> 4;
  const int cl = lane & 15;
  float* of = (float*)out;
  u16* ob = (u16*)out;
#pragma unroll
  for (int mt = 0; mt < 4; ++mt) {
#pragma unroll
    for (int nt = 0; nt < 4; ++nt) {
      int col = bn + wn + nt * 16 + cl;
      float bv = bias[col];
#pragma unroll
      for (int i = 0; i < 8; ++i) {
        int row = bm + wm + mt * 16 + i + 8 * hh;
        float v = (acc[mt][nt][i] + bv) * alpha;
        if (out_bf16) ob[(size_t)row * N + col] = f2bf(v);
        else          of[(size_t)row * N + col] = v;
      }
    }
  }
}

// ---------------------------------------------------------------------------
// Flash attention (causal, GQA). One block = (batch b, head h, 128 q rows).
// 8 waves x 16 q rows. K staged row-major; V staged TRANSPOSED (Vt[d][keys])
// so every P*V B-fragment is a contiguous ds_load_b128 pair instead of 16
// strided ds_load_u16. Q pre-scaled by 1/sqrt(d) in the Q projection.
// ---------------------------------------------------------------------------
#define KSTR 136  // 128 + 8 pad (K tile rows)
#define VSTR 40   // 32 + 8 pad  (Vt rows = d columns, 32 keys each)
#define PSTR 40

__global__ __launch_bounds__(256) void k_flash_attn(const u16* __restrict__ Q,
                                                    const u16* __restrict__ K,
                                                    const u16* __restrict__ V,
                                                    u16* __restrict__ Y) {
  __shared__ u16 Ks[32 * KSTR];       // 32 keys x 128 d
  __shared__ u16 Vt[128 * VSTR];      // 128 d x 32 keys (transposed)
  __shared__ u16 Ps[8 * 16 * PSTR];   // per-wave 16x32 P tile

  const int tid  = threadIdx.x;
  const int lane = tid & 31;
  const int w    = tid >> 5;
  const int hh   = lane >> 4;
  const int cl   = lane & 15;

  const int q0 = blockIdx.x * 128;
  const int h  = blockIdx.y;
  const int b  = blockIdx.z;
  const int g  = h >> 2;              // kv group (GROUP = H/G = 4)
  const int qr = q0 + w * 16;         // first q row owned by this wave

  // Q fragments: 16 rows x 128 (4 chunks of 32), straight from global.
  v16u aq[4];
#pragma unroll
  for (int c = 0; c < 4; ++c)
    aq[c] = load_frag_16x32(Q + (size_t)(b * T_ + qr) * D_ + h * d_ + c * 32, D_);

  v8f o[8];
#pragma unroll
  for (int n = 0; n < 8; ++n) o[n] = vzero8();
  float mrow[8], lrow[8];
#pragma unroll
  for (int i = 0; i < 8; ++i) { mrow[i] = -3.0e38f; lrow[i] = 0.0f; }

  u16* Pw = Ps + w * 16 * PSTR;
  const int ktiles = (q0 + 128) / 32;

  for (int kt = 0; kt < ktiles; ++kt) {
    // Stage K row-major and V transposed. 32 rows x 128 cols each.
#pragma unroll
    for (int j = 0; j < 2; ++j) {
      int idx = tid + j * 256;        // 0..511
      int row = idx >> 4;             // key 0..31
      int seg = (idx & 15) * 8;       // d chunk
      size_t src = (size_t)(b * T_ + kt * 32 + row) * KV_ + g * d_ + seg;
      *(v8u*)(Ks + row * KSTR + seg) = *(const v8u*)(K + src);
      v8u vv = *(const v8u*)(V + src);
#pragma unroll
      for (int e = 0; e < 8; ++e)     // scatter: Vt[dcol][key]
        Vt[(seg + e) * VSTR + row] = vv[e];
    }
    __syncthreads();

    if (kt * 32 <= qr + 15) {         // causal: tile touches our rows
      // S = Q * K^T  (two 16-key tiles, K-dim = d = 128)
      v8f s0 = vzero8(), s1 = vzero8();
#pragma unroll
      for (int c = 0; c < 4; ++c) {
        v16u bk0 = load_frag_16x32(Ks + 0 * KSTR + c * 32, KSTR);
        v16u bk1 = load_frag_16x32(Ks + 16 * KSTR + c * 32, KSTR);
        s0 = wmma_bf16(aq[c], bk0, s0);
        s1 = wmma_bf16(aq[c], bk1, s1);
      }

      // Online softmax per row (one row = 16 lanes of a half-wave).
#pragma unroll
      for (int i = 0; i < 8; ++i) {
        int rq  = qr + i + 8 * hh;
        int k0c = kt * 32 + cl;
        float x0 = (k0c      > rq) ? -1.0e30f : s0[i];
        float x1 = (k0c + 16 > rq) ? -1.0e30f : s1[i];
        float tm = rowmax16(fmaxf(x0, x1));
        float nm = fmaxf(mrow[i], tm);
        float corr = __expf(mrow[i] - nm);
        mrow[i] = nm;
        float p0 = __expf(x0 - nm);
        float p1 = __expf(x1 - nm);
        lrow[i] = lrow[i] * corr + rowsum16(p0 + p1);
#pragma unroll
        for (int n = 0; n < 8; ++n) o[n][i] *= corr;
        Pw[(i + 8 * hh) * PSTR + cl]      = f2bf(p0);
        Pw[(i + 8 * hh) * PSTR + 16 + cl] = f2bf(p1);
      }
      asm volatile("s_wait_dscnt 0" ::: "memory");  // wave-local P in LDS

      // O += P * V: A = P (16x32 keys), B = Vt rows (16 d-cols x 32 keys).
      v16u pf = load_frag_16x32(Pw, PSTR);
#pragma unroll
      for (int n = 0; n < 8; ++n) {
        v16u vf = load_frag_16x32(Vt + (n * 16) * VSTR, VSTR);
        o[n] = wmma_bf16(pf, vf, o[n]);
      }
    }
    __syncthreads();
  }

  // Normalize and write Y[b, t, h*d + col] as bf16 for the output projection.
#pragma unroll
  for (int i = 0; i < 8; ++i) {
    float inv = 1.0f / lrow[i];
    size_t row = (size_t)(b * T_ + qr + i + 8 * hh);
#pragma unroll
    for (int n = 0; n < 8; ++n)
      Y[row * D_ + h * d_ + n * 16 + cl] = f2bf(o[n][i] * inv);
  }
}

// ---------------------------------------------------------------------------
// Host-side orchestration
// ---------------------------------------------------------------------------
extern "C" void kernel_launch(void* const* d_in, const int* in_sizes, int n_in,
                              void* d_out, int out_size, void* d_ws, size_t ws_size,
                              hipStream_t stream) {
  (void)in_sizes; (void)n_in; (void)out_size; (void)ws_size;

  const float* hs = (const float*)d_in[0];
  const float* Wq = (const float*)d_in[1];
  const float* bq = (const float*)d_in[2];
  const float* Wk = (const float*)d_in[3];
  const float* bk = (const float*)d_in[4];
  const float* Wv = (const float*)d_in[5];
  const float* bv = (const float*)d_in[6];
  const float* Wo = (const float*)d_in[7];
  const float* bo = (const float*)d_in[8];

  const size_t NX  = (size_t)B_ * T_ * D_;   // 16,777,216
  const size_t NWQ = (size_t)D_ * D_;
  const size_t NWK = (size_t)KV_ * D_;
  const size_t NKV = (size_t)B_ * T_ * KV_;

  u16* p = (u16*)d_ws;
  u16* Xb  = p; p += NX;
  u16* Wqb = p; p += NWQ;
  u16* Wkb = p; p += NWK;
  u16* Wvb = p; p += NWK;
  u16* Wob = p; p += NWQ;
  u16* Qb  = p; p += NX;
  u16* Kb  = p; p += NKV;
  u16* Vb  = p; p += NKV;
  u16* Yb  = p; p += NX;

  // 1) fp32 -> bf16 conversions
  k_f32_to_bf16<<<2048, 256, 0, stream>>>(hs, Xb, NX);
  k_f32_to_bf16<<<2048, 256, 0, stream>>>(Wq, Wqb, NWQ);
  k_f32_to_bf16<<<1024, 256, 0, stream>>>(Wk, Wkb, NWK);
  k_f32_to_bf16<<<1024, 256, 0, stream>>>(Wv, Wvb, NWK);
  k_f32_to_bf16<<<2048, 256, 0, stream>>>(Wo, Wob, NWQ);

  const int M = B_ * T_;                        // 4096
  const float scale = 0.08838834764831845f;     // 1/sqrt(128)

  // 2) projections (Q carries the 1/sqrt(d) scale)
  k_gemm_xwt<<<dim3(D_ / GT_N, M / GT_M), 256, 0, stream>>>(
      Xb, Wqb, bq, Qb, M, D_, D_, scale, /*bf16 out*/1);
  k_gemm_xwt<<<dim3(KV_ / GT_N, M / GT_M), 256, 0, stream>>>(
      Xb, Wkb, bk, Kb, M, KV_, D_, 1.0f, 1);
  k_gemm_xwt<<<dim3(KV_ / GT_N, M / GT_M), 256, 0, stream>>>(
      Xb, Wvb, bv, Vb, M, KV_, D_, 1.0f, 1);

  // 3) causal GQA flash attention
  k_flash_attn<<<dim3(T_ / 128, H_, B_), 256, 0, stream>>>(Qb, Kb, Vb, Yb);

  // 4) output projection -> fp32 result
  k_gemm_xwt<<<dim3(D_ / GT_N, M / GT_M), 256, 0, stream>>>(
      Yb, Wob, bo, (float*)d_out, M, D_, D_, 1.0f, /*fp32 out*/0);
}
